// VectorQuantizer_10050223473099
// MI455X (gfx1250) — matrix operand order repaired and checked by reading, not compiled, not run on hardware
//
#include <hip/hip_runtime.h>
#include <hip/hip_bf16.h>

// ---------------------------------------------------------------------------
// VQ-VAE vector quantizer, fused for MI455X (gfx1250, wave32, WMMA).
//   K0: codebook f32->bf16 + exact ||e||^2
//   K1: fused (z . e) GEMM via v_wmma_f32_16x16x32_bf16 + running argmin,
//       LDS-staged codebook chunks (double-buffered async-to-LDS),
//       sched_group_barrier-pinned DS/WMMA software pipeline.
//   K2: exact f32 gather z_q + per-block loss partials
//   K3: deterministic loss reduction
// ---------------------------------------------------------------------------

typedef __attribute__((ext_vector_type(16))) __bf16 v16bf;
typedef __attribute__((ext_vector_type(8)))  float  v8f;
typedef __attribute__((ext_vector_type(4)))  int    v4i;

#define N_TOKENS  32768
#define NUM_EMB   8192
#define DIM       256
#define ZQ_ELEMS  8388608   // N_TOKENS * DIM
#define CHUNK     16        // codebook rows per sweep step
#define N_CHUNKS  512       // NUM_EMB / CHUNK
#define CHUNK_ELEMS (CHUNK * DIM)   // 4096 bf16 = 8 KB
#define LDS_ROW   264       // 256 + 8 pad -> 528B row stride (16B aligned, bank-clean)
#define GATHER_BLOCKS 8192  // ZQ_ELEMS / (256 threads * 4 elems)

#define AS1 __attribute__((address_space(1)))
#define AS3 __attribute__((address_space(3)))

#if __has_builtin(__builtin_amdgcn_global_load_async_to_lds_b128)
#define HAVE_ASYNC_LDS 1
#else
#define HAVE_ASYNC_LDS 0
#endif

// scheduling directives (best-effort; no-op if unavailable)
#if __has_builtin(__builtin_amdgcn_sched_group_barrier)
#define SCHED_GROUP(mask, n, id) __builtin_amdgcn_sched_group_barrier((mask), (n), (id))
#else
#define SCHED_GROUP(mask, n, id)
#endif
#define SG_WMMA   0x008   // MFMA/WMMA group
#define SG_DSREAD 0x100   // DS read group

__device__ __forceinline__ void wait_async0() {
#if __has_builtin(__builtin_amdgcn_s_wait_asynccnt)
  __builtin_amdgcn_s_wait_asynccnt(0);
#else
  asm volatile("s_wait_asynccnt 0" ::: "memory");
#endif
}

__device__ __forceinline__ unsigned short f2bf(float f) {
  // round-to-nearest-even f32 -> bf16
  unsigned int u = __float_as_uint(f);
  u += 0x7FFFu + ((u >> 16) & 1u);
  return (unsigned short)(u >> 16);
}

union Frag16 { unsigned short u[16]; uint4 q[2]; v16bf v; };

// ---------------------------------------------------------------------------
// K0: codebook prep: bf16 copy + exact squared norms
// grid: NUM_EMB blocks x 256 threads (one row per block)
// ---------------------------------------------------------------------------
__global__ void __launch_bounds__(256) vq_prep_cb(const float* __restrict__ cb,
                                                  unsigned short* __restrict__ cb_bf,
                                                  float* __restrict__ enorm) {
  __shared__ float red[256];
  const int row = blockIdx.x;
  const int tid = threadIdx.x;
  float v = cb[(size_t)row * DIM + tid];
  cb_bf[(size_t)row * DIM + tid] = f2bf(v);
  red[tid] = v * v;
  __syncthreads();
  for (int s = 128; s > 0; s >>= 1) {
    if (tid < s) red[tid] += red[tid + s];
    __syncthreads();
  }
  if (tid == 0) enorm[row] = red[0];
}

// ---------------------------------------------------------------------------
// K1: fused distance + argmin.
//   block = 128 threads = 4 waves; wave owns 32 tokens (two 16-row A sets);
//   block stages each 8KB codebook chunk in LDS once (double-buffered).
// dist(m,k) = ||e_k||^2 - 2*dot(z_m,e_k)  (row-constant ||z||^2 dropped)
// grid: 256 blocks x 128 threads
// ---------------------------------------------------------------------------
__global__ void __launch_bounds__(128, 1)
vq_argmin(const float* __restrict__ z,
          const unsigned short* __restrict__ cb_bf,
          const float* __restrict__ enorm,
          int* __restrict__ out_idx) {
  __shared__ unsigned short lds_b[2][CHUNK * LDS_ROW];   // 2 x 8.25 KB

  const int tid  = threadIdx.x;
  const int lane = tid & 31;
  const int wave = tid >> 5;
  const int tok0 = blockIdx.x * 128 + wave * 32;
  const int nloc = lane & 15;
  const int half = lane >> 4;

  // ---- Build both A fragment sets once (ISA 7.12.2 16x32 bf16 layout) ----
  Frag16 a0[8], a1[8];
  {
    const float* r0 = z + (size_t)(tok0 + nloc) * DIM;
    const float* r1 = z + (size_t)(tok0 + 16 + nloc) * DIM;
#pragma unroll
    for (int f = 0; f < 8; ++f) {
      const float* p00 = r0 + f * 32 + half * 8;
      const float* p01 = r0 + f * 32 + 16 + half * 8;
      const float* p10 = r1 + f * 32 + half * 8;
      const float* p11 = r1 + f * 32 + 16 + half * 8;
#pragma unroll
      for (int j = 0; j < 8; ++j) {
        a0[f].u[j]     = f2bf(p00[j]);
        a0[f].u[8 + j] = f2bf(p01[j]);
        a1[f].u[j]     = f2bf(p10[j]);
        a1[f].u[8 + j] = f2bf(p11[j]);
      }
    }
  }

  float best0[8], best1[8];
  int   bidx0[8], bidx1[8];
#pragma unroll
  for (int g = 0; g < 8; ++g) {
    best0[g] = 3.4e38f; bidx0[g] = 0;
    best1[g] = 3.4e38f; bidx1[g] = 0;
  }

  // thread's slice of an 8KB chunk: 4 x 16B pieces q = tid + i*128
  // piece q -> global offset q*16B ; LDS offset (q>>5)*LDS_ROW + (q&31)*8 ushorts
  // ---- prologue: stage chunk 0 into buffer 0 ----
  {
    const char* src = (const char*)cb_bf;   // chunk 0 is contiguous 8KB
#if HAVE_ASYNC_LDS
#pragma unroll
    for (int i = 0; i < 4; ++i) {
      const int q = tid + i * 128;
      __builtin_amdgcn_global_load_async_to_lds_b128(
          (AS1 v4i*)(src + (size_t)q * 16),
          (AS3 v4i*)&lds_b[0][(q >> 5) * LDS_ROW + (q & 31) * 8], 0, 0);
    }
    wait_async0();
#else
#pragma unroll
    for (int i = 0; i < 4; ++i) {
      const int q = tid + i * 128;
      uint4 v = *(const uint4*)(src + (size_t)q * 16);
      *(uint4*)&lds_b[0][(q >> 5) * LDS_ROW + (q & 31) * 8] = v;
    }
#endif
  }
  __syncthreads();

#pragma unroll 1
  for (int c = 0; c < N_CHUNKS; ++c) {
    // ---- stage chunk c+1 into the other buffer (overlaps the WMMAs).
    // c==511 stages a dummy chunk from the enorm region (valid ws memory).
    const char* nsrc = (const char*)(cb_bf + (size_t)(c + 1) * CHUNK_ELEMS);
    unsigned short* nbuf = lds_b[(c + 1) & 1];
#if HAVE_ASYNC_LDS
#pragma unroll
    for (int i = 0; i < 4; ++i) {
      const int q = tid + i * 128;
      __builtin_amdgcn_global_load_async_to_lds_b128(
          (AS1 v4i*)(nsrc + (size_t)q * 16),
          (AS3 v4i*)&nbuf[(q >> 5) * LDS_ROW + (q & 31) * 8], 0, 0);
    }
#else
    uint4 nv[4];
#pragma unroll
    for (int i = 0; i < 4; ++i)
      nv[i] = *(const uint4*)(nsrc + (size_t)(tid + i * 128) * 16);
#endif

    // hoisted: ||e||^2 load latency overlaps the WMMA chain below
    const int k = c * CHUNK + nloc;
    const float en = enorm[k];

    // ---- 16 WMMAs, software-pipelined B fragments from LDS ----
    const unsigned short* bb = &lds_b[c & 1][nloc * LDS_ROW + half * 8];
    Frag16 bcur, bnxt;
    bcur.q[0] = *(const uint4*)(bb);
    bcur.q[1] = *(const uint4*)(bb + 16);
    v8f acc0 = {}, acc1 = {};
#pragma unroll
    for (int f = 0; f < 8; ++f) {
      if (f < 7) {
        bnxt.q[0] = *(const uint4*)(bb + (f + 1) * 32);
        bnxt.q[1] = *(const uint4*)(bb + (f + 1) * 32 + 16);
      }
      acc0 = __builtin_amdgcn_wmma_f32_16x16x32_bf16(
          false, a0[f].v, false, bcur.v, (short)0, acc0, false, false);
      acc1 = __builtin_amdgcn_wmma_f32_16x16x32_bf16(
          false, a1[f].v, false, bcur.v, (short)0, acc1, false, false);
      if (f < 7) bcur = bnxt;
    }

    // ---- running per-lane argmin for both token sets ----
#pragma unroll
    for (int g = 0; g < 8; ++g) {
      float d0 = fmaf(-2.0f, acc0[g], en);
      float d1 = fmaf(-2.0f, acc1[g], en);
      if (d0 < best0[g]) { best0[g] = d0; bidx0[g] = k; }
      if (d1 < best1[g]) { best1[g] = d1; bidx1[g] = k; }
    }

    // ---- pin the DS-read / WMMA interleave so B loads run one stage ahead:
    //   DS x4 | (WMMA x2, DS x2) x6 | WMMA x2 | WMMA x2
    SCHED_GROUP(SG_DSREAD, 4, 0);
#pragma unroll
    for (int i = 0; i < 6; ++i) {
      SCHED_GROUP(SG_WMMA, 2, 0);
      SCHED_GROUP(SG_DSREAD, 2, 0);
    }
    SCHED_GROUP(SG_WMMA, 2, 0);
    SCHED_GROUP(SG_WMMA, 2, 0);

#if HAVE_ASYNC_LDS
    wait_async0();                 // next buffer fully in LDS before publish
#else
#pragma unroll
    for (int i = 0; i < 4; ++i) {
      const int q = tid + i * 128;
      *(uint4*)&nbuf[(q >> 5) * LDS_ROW + (q & 31) * 8] = nv[i];
    }
#endif
    __syncthreads();               // publish buf[(c+1)&1]; retire buf[c&1]
  }

  // ---- cross-lane argmin within each 16-lane half, then write indices ----
#pragma unroll
  for (int g = 0; g < 8; ++g) {
#pragma unroll
    for (int off = 8; off >= 1; off >>= 1) {
      float ov0 = __shfl_xor(best0[g], off, 32);
      int   oi0 = __shfl_xor(bidx0[g], off, 32);
      if (ov0 < best0[g] || (ov0 == best0[g] && oi0 < bidx0[g])) { best0[g] = ov0; bidx0[g] = oi0; }
      float ov1 = __shfl_xor(best1[g], off, 32);
      int   oi1 = __shfl_xor(bidx1[g], off, 32);
      if (ov1 < best1[g] || (ov1 == best1[g] && oi1 < bidx1[g])) { best1[g] = ov1; bidx1[g] = oi1; }
    }
  }
  if (nloc == 0) {
#pragma unroll
    for (int g = 0; g < 8; ++g) {
      out_idx[tok0 + g + 8 * half]      = bidx0[g];   // set0 token m = g + 8*half
      out_idx[tok0 + 16 + g + 8 * half] = bidx1[g];   // set1 token m+16
    }
  }
}

// ---------------------------------------------------------------------------
// K2: exact f32 gather z_q = codebook[idx], write z_q + indices(as float),
//     per-block partial sums of (z_q - z)^2.
// grid: 8192 blocks x 256 threads, 4 elems/thread (exact cover of 8.4M)
// ---------------------------------------------------------------------------
__global__ void __launch_bounds__(256) vq_gather(const float* __restrict__ z,
                                                 const float* __restrict__ cb,
                                                 const int* __restrict__ idx,
                                                 float* __restrict__ out,
                                                 float* __restrict__ partials) {
  __shared__ float red[256];
  const int tid = threadIdx.x;
  const size_t t = (size_t)blockIdx.x * 256 + tid;
  const size_t e = t * 4;
  const int token = (int)(e >> 8);
  const int d = (int)(e & 255);
  const int k = idx[token];

  float4 zv = *(const float4*)(z + e);
  float4 qv = *(const float4*)(cb + (size_t)k * DIM + d);
  *(float4*)(out + e) = qv;                    // forward z_q_st == z_q

  float dx = qv.x - zv.x, dy = qv.y - zv.y;
  float dz = qv.z - zv.z, dw = qv.w - zv.w;
  red[tid] = dx * dx + dy * dy + dz * dz + dw * dw;

  if (d == 0) out[(size_t)ZQ_ELEMS + 1 + token] = (float)k;  // indices tail

  __syncthreads();
  for (int s = 128; s > 0; s >>= 1) {
    if (tid < s) red[tid] += red[tid + s];
    __syncthreads();
  }
  if (tid == 0) partials[blockIdx.x] = red[0];
}

// ---------------------------------------------------------------------------
// K3: deterministic final reduction -> loss = 1.25 * MSE
// ---------------------------------------------------------------------------
__global__ void __launch_bounds__(256) vq_finalize(const float* __restrict__ partials,
                                                   float* __restrict__ out) {
  __shared__ float red[256];
  const int tid = threadIdx.x;
  float s = 0.0f;
  for (int i = tid; i < GATHER_BLOCKS; i += 256) s += partials[i];
  red[tid] = s;
  __syncthreads();
  for (int r = 128; r > 0; r >>= 1) {
    if (tid < r) red[tid] += red[tid + r];
    __syncthreads();
  }
  if (tid == 0) out[ZQ_ELEMS] = 1.25f * red[0] / (float)ZQ_ELEMS;
}

// ---------------------------------------------------------------------------
extern "C" void kernel_launch(void* const* d_in, const int* in_sizes, int n_in,
                              void* d_out, int out_size, void* d_ws, size_t ws_size,
                              hipStream_t stream) {
  const float* z  = (const float*)d_in[0];   // [32768, 256] f32
  const float* cb = (const float*)d_in[1];   // [8192, 256]  f32
  float* out = (float*)d_out;                // [z_q (8388608) | loss (1) | idx (32768)]

  // workspace layout (16B-aligned slices)
  char* w = (char*)d_ws;
  unsigned short* cb_bf    = (unsigned short*)(w);                // 4,194,304 B
  float*          enorm    = (float*)(w + 4194304);               //    32,768 B
  int*            idx      = (int*)(w + 4227072);                 //   131,072 B
  float*          partials = (float*)(w + 4358144);               //    32,768 B

  vq_prep_cb<<<NUM_EMB, 256, 0, stream>>>(cb, cb_bf, enorm);
  vq_argmin<<<N_TOKENS / 128, 128, 0, stream>>>(z, cb_bf, enorm, idx);
  vq_gather<<<GATHER_BLOCKS, 256, 0, stream>>>(z, cb, idx, out, partials);
  vq_finalize<<<1, 256, 0, stream>>>(partials, out);
}
